// SetPartitionTri_17351667876289
// MI455X (gfx1250) — 16-bit and 8-bit matrix operand order repaired and checked by reading, compiled, not run on hardware
//
#include <hip/hip_runtime.h>
#include <hip/hip_bf16.h>
#include <stdint.h>

typedef __attribute__((ext_vector_type(16))) __bf16 v16bf;
typedef __attribute__((ext_vector_type(8)))  float  v8f;
typedef __attribute__((ext_vector_type(16))) int    v16i;

#define BATCH 8
#define NPTS  2048
#define CIN   128

#define FP8_ONE  0x38u
#define FP8_ZERO 0x00u

#if defined(__gfx1250__) && __has_builtin(__builtin_amdgcn_global_load_async_to_lds_b128) && __has_builtin(__builtin_amdgcn_s_wait_asynccnt)
#define USE_ASYNC_LDS 1
typedef int v4i_vs __attribute__((vector_size(16)));
typedef __attribute__((address_space(1))) v4i_vs* as1_v4i;
typedef __attribute__((address_space(3))) v4i_vs* as3_v4i;
#else
#define USE_ASYNC_LDS 0
#endif

union bfu { __bf16 b; unsigned short s; };
__device__ inline __bf16 bits2bf(unsigned short s) { bfu u; u.s = s; return u.b; }
__device__ inline unsigned short f2bfbits(float f) {
    unsigned u = __float_as_uint(f);
    unsigned r = (u + 0x7FFFu + ((u >> 16) & 1u)) >> 16;
    return (unsigned short)r;
}
__device__ inline float bf2f(unsigned short s) { return __uint_as_float(((unsigned)s) << 16); }

__global__ void cvt_bf16_kernel(const float* __restrict__ in, unsigned short* __restrict__ out, int n) {
    int i = blockIdx.x * blockDim.x + threadIdx.x;
    if (i < n) out[i] = f2bfbits(in[i]);
}

__global__ void init_kernel(int* flags, float* loss) {
    if (threadIdx.x == 0 && blockIdx.x == 0) { flags[0] = 0; flags[1] = 0; *loss = 0.0f; }
}

__global__ void gemm_bf16_kernel(const unsigned short* __restrict__ A,
                                 const unsigned short* __restrict__ W,
                                 const float* __restrict__ bias,
                                 unsigned short* __restrict__ C,
                                 int M, int N, int K, int relu)
{
    int wave = (int)((blockIdx.x * blockDim.x + threadIdx.x) >> 5);
    int lane = threadIdx.x & 31;
    int ntN = N >> 4;
    int ntiles = (M >> 4) * ntN;
    if (wave >= ntiles) return;
    int tm = wave / ntN, tn = wave % ntN;
    int m = lane & 15, h = lane >> 4;
    int row  = tm * 16 + m;
    int ncol = tn * 16 + m;
    v8f acc = {};
    for (int kk = 0; kk < K; kk += 32) {
        v16bf a, b;
#pragma unroll
        for (int v = 0; v < 8; ++v) {
            int ka = kk + ((v >> 2) << 4) + (h << 3) + ((v & 3) << 1);
            a[2 * v]     = bits2bf(A[(size_t)row * K + ka]);
            a[2 * v + 1] = bits2bf(A[(size_t)row * K + ka + 1]);
            int kb = kk + ((v >> 2) << 4) + (h << 3) + ((v & 3) << 1);
            b[2 * v]     = bits2bf(W[(size_t)ncol * K + kb]);
            b[2 * v + 1] = bits2bf(W[(size_t)ncol * K + kb + 1]);
        }
        acc = __builtin_amdgcn_wmma_f32_16x16x32_bf16(false, a, false, b, (short)0, acc, false, false);
    }
#pragma unroll
    for (int r = 0; r < 8; ++r) {
        int gi = tm * 16 + (h << 3) + r;
        int gj = tn * 16 + m;
        float val = acc[r] + bias[gj];
        if (relu) val = fmaxf(val, 0.0f);
        C[(size_t)gi * N + gj] = f2bfbits(val);
    }
}

__global__ void rownorm_kernel(const unsigned short* __restrict__ U, float* __restrict__ sq, int rows) {
    int i = blockIdx.x * blockDim.x + threadIdx.x;
    if (i >= rows) return;
    const unsigned short* p = U + (size_t)i * 64;
    float s = 0.f;
#pragma unroll
    for (int k = 0; k < 64; ++k) { float v = bf2f(p[k]); s += v * v; }
    sq[i] = s;
}

__global__ void gram_adj_kernel(const unsigned short* __restrict__ U,
                                const float* __restrict__ sq,
                                unsigned char* __restrict__ adj)
{
    int wave = (int)((blockIdx.x * blockDim.x + threadIdx.x) >> 5);
    int lane = threadIdx.x & 31;
    const int NT = NPTS / 16;
    int total = BATCH * NT * NT;
    if (wave >= total) return;
    int b   = wave / (NT * NT);
    int rem = wave % (NT * NT);
    int tm = rem / NT, tn = rem % NT;
    const unsigned short* Ub = U + (size_t)b * NPTS * 64;
    const float* sqb = sq + (size_t)b * NPTS;
    unsigned char* adjb = adj + (size_t)b * NPTS * NPTS;
    int m = lane & 15, h = lane >> 4;
    v8f acc = {};
    for (int kk = 0; kk < 64; kk += 32) {
        v16bf a, bf;
#pragma unroll
        for (int v = 0; v < 8; ++v) {
            int ka = kk + ((v >> 2) << 4) + (h << 3) + ((v & 3) << 1);
            a[2 * v]     = bits2bf(Ub[(size_t)(tm * 16 + m) * 64 + ka]);
            a[2 * v + 1] = bits2bf(Ub[(size_t)(tm * 16 + m) * 64 + ka + 1]);
            int kb = kk + ((v >> 2) << 4) + (h << 3) + ((v & 3) << 1);
            bf[2 * v]     = bits2bf(Ub[(size_t)(tn * 16 + m) * 64 + kb]);
            bf[2 * v + 1] = bits2bf(Ub[(size_t)(tn * 16 + m) * 64 + kb + 1]);
        }
        acc = __builtin_amdgcn_wmma_f32_16x16x32_bf16(false, a, false, bf, (short)0, acc, false, false);
    }
#pragma unroll
    for (int r = 0; r < 8; ++r) {
        int i = tm * 16 + (h << 3) + r;
        int j = tn * 16 + m;
        float d = fmaxf(sqb[i] + sqb[j] - 2.0f * acc[r], 0.0f);
        adjb[(size_t)i * NPTS + j] = (d <= 1.0f || i == j) ? FP8_ONE : FP8_ZERO;
    }
}


__device__ inline void stage_B_tile(unsigned char* lds, const unsigned char* srcb,
                                    int n0, int kk, int tid)
{
#if USE_ASYNC_LDS
#pragma unroll
    for (int j = 0; j < 2; ++j) {
        int chunk = j * 256 + tid;
        int n = chunk >> 3, c16 = (chunk & 7) << 4;
        const unsigned char* g = srcb + (size_t)(n0 + n) * NPTS + kk + c16;
        unsigned char* l = lds + n * 128 + c16;
        __builtin_amdgcn_global_load_async_to_lds_b128(
            (as1_v4i)(uintptr_t)g, (as3_v4i)(void*)l, 0, 0);
    }
#else
    for (int t = tid; t < 2048; t += 256) {
        int n = t >> 5, c4 = (t & 31) << 2;
        *(unsigned int*)(lds + n * 128 + c4) =
            *(const unsigned int*)(srcb + (size_t)(n0 + n) * NPTS + kk + c4);
    }
#endif
}

__device__ inline void wait_stage() {
#if USE_ASYNC_LDS
    __builtin_amdgcn_s_wait_asynccnt(0);
#endif
    __syncthreads();
}

__global__ void bool_square_kernel(const unsigned char* __restrict__ src,
                                   unsigned char* __restrict__ dst,
                                   int* __restrict__ changed,
                                   const int* __restrict__ converged)
{
    const int MB = 128, NB = 64;
    const int blocksPerBatch = (NPTS / MB) * (NPTS / NB);
    int gb = blockIdx.x;
    if (gb >= BATCH * blocksPerBatch) return;
    int b  = gb / blocksPerBatch;
    int rb = gb % blocksPerBatch;
    int m0 = (rb / (NPTS / NB)) * MB;
    int n0 = (rb % (NPTS / NB)) * NB;
    const unsigned char* srcb = src + (size_t)b * NPTS * NPTS;
    unsigned char* dstb = dst + (size_t)b * NPTS * NPTS;
    int tid = threadIdx.x;

    if (*converged) {
        for (int t = tid; t < MB * NB / 4; t += 256) {
            int r = t >> 4, c4 = (t & 15) << 2;
            *(unsigned int*)(dstb + (size_t)(m0 + r) * NPTS + n0 + c4) =
                *(const unsigned int*)(srcb + (size_t)(m0 + r) * NPTS + n0 + c4);
        }
        return;
    }

    __shared__ __align__(16) unsigned char Bs[2][NB * 128];
    __shared__ int chg;
    if (tid == 0) chg = 0;

    int waveId = tid >> 5, lane = tid & 31;
    int m = lane & 15, h = lane >> 4;
    int rowA = m0 + waveId * 16 + m;
    v8f acc[4] = {v8f{}, v8f{}, v8f{}, v8f{}};

    stage_B_tile(Bs[0], srcb, n0, 0, tid);
    wait_stage();

    int cur = 0;
    for (int kk = 0; kk < NPTS; kk += 128) {
        if (kk + 128 < NPTS)
            stage_B_tile(Bs[cur ^ 1], srcb, n0, kk + 128, tid);

        v16i a;
#pragma unroll
        for (int v = 0; v < 16; ++v) {
            int half = v >> 3, v8 = v & 7;
            int ka = (half << 6) + ((v8 >> 1) << 4) + (h << 3) + ((v8 & 1) << 2);
            a[v] = *(const int*)(srcb + (size_t)rowA * NPTS + kk + ka);
        }
        if (kk + 128 < NPTS)
            __builtin_prefetch(srcb + (size_t)rowA * NPTS + kk + 128, 0, 0);

#pragma unroll
        for (int q = 0; q < 4; ++q) {
            v16i bb;
#pragma unroll
            for (int v = 0; v < 16; ++v) {
                int kb = ((v >> 3) << 6) + (((v & 7) >> 1) << 4) + (h << 3) + ((v & 1) << 2);
                bb[v] = *(const int*)(Bs[cur] + (q * 16 + m) * 128 + kb);
            }
            acc[q] = __builtin_amdgcn_wmma_f32_16x16x128_fp8_fp8(a, bb, (short)0, acc[q], false, false);
        }

        wait_stage();
        cur ^= 1;
    }

    int wchg = 0;
#pragma unroll
    for (int q = 0; q < 4; ++q) {
#pragma unroll
        for (int r = 0; r < 8; ++r) {
            int gi = m0 + waveId * 16 + (h << 3) + r;
            int gj = n0 + q * 16 + m;
            unsigned char nv = (acc[q][r] > 0.5f) ? FP8_ONE : FP8_ZERO;
            unsigned char ov = srcb[(size_t)gi * NPTS + gj];
            wchg |= (nv != ov);
            dstb[(size_t)gi * NPTS + gj] = nv;
        }
    }
    if (wchg) atomicOr(&chg, 1);
    __syncthreads();
    if (tid == 0 && chg) atomicOr(changed, 1);
}

__global__ void closure_flag_kernel(int* changed, int* converged) {
    if (threadIdx.x == 0 && blockIdx.x == 0) {
        if (*changed == 0) *converged = 1;
        *changed = 0;
    }
}

__device__ inline float pair_dist(const unsigned short* Ub, const float* sqb, int i, int j) {
    const unsigned short* pi = Ub + (size_t)i * 64;
    const unsigned short* pj = Ub + (size_t)j * 64;
    float dot = 0.f;
#pragma unroll
    for (int k = 0; k < 64; ++k) dot += bf2f(pi[k]) * bf2f(pj[k]);
    return fmaxf(sqb[i] + sqb[j] - 2.0f * dot, 0.0f);
}

__global__ void triplet_kernel(const int* __restrict__ labels, const int* __restrict__ tri,
                               const unsigned short* __restrict__ U, const float* __restrict__ sq,
                               float* __restrict__ loss)
{
    int b = blockIdx.x;
    const int* lab = labels + (size_t)b * NPTS;
    const unsigned short* Ub = U + (size_t)b * NPTS * 64;
    const float* sqb = sq + (size_t)b * NPTS;
    __shared__ float s_sum[256];
    __shared__ int   s_cnt[256];
    __shared__ int   s_max[256];
    int tid = threadIdx.x;
    int mx = 0;
    for (int i = tid; i < NPTS; i += 256) mx = max(mx, lab[i]);
    float ssum = 0.f; int scnt = 0;
    for (int t = tid; t < 200; t += 256) {
        int a = tri[t * 3 + 0], p = tri[t * 3 + 1], n = tri[t * 3 + 2];
        bool valid = (a != p) && (lab[a] == lab[p]) && (lab[p] != lab[n]);
        if (valid) {
            float dap = pair_dist(Ub, sqb, a, p);
            float dan = pair_dist(Ub, sqb, a, n);
            ssum += fmaxf(dap - dan + 2.0f, 0.0f);
            scnt += 1;
        }
    }
    s_sum[tid] = ssum; s_cnt[tid] = scnt; s_max[tid] = mx;
    __syncthreads();
    for (int s = 128; s > 0; s >>= 1) {
        if (tid < s) {
            s_sum[tid] += s_sum[tid + s];
            s_cnt[tid] += s_cnt[tid + s];
            s_max[tid] = max(s_max[tid], s_max[tid + s]);
        }
        __syncthreads();
    }
    if (tid == 0) {
        int cnt = s_cnt[0];
        bool use = (s_max[0] > 0) && (cnt > 0);
        float m = s_sum[0] / (float)max(cnt, 1);
        atomicAdd(loss, use ? m : 0.0f);
    }
}

__global__ void clusters_kernel(const unsigned char* __restrict__ adj, float* __restrict__ out) {
    int g = blockIdx.x * blockDim.x + threadIdx.x;
    if (g >= BATCH * NPTS) return;
    int b = g >> 11, i = g & (NPTS - 1);
    const unsigned char* ab = adj + (size_t)b * NPTS * NPTS;
    int best = -1;
    for (int j = 0; j < NPTS; ++j)
        if (ab[(size_t)j * NPTS + i]) best = j;
    out[g] = (float)best;
}

extern "C" void kernel_launch(void* const* d_in, const int* in_sizes, int n_in,
                              void* d_out, int out_size, void* d_ws, size_t ws_size,
                              hipStream_t stream) {
    const float* x      = (const float*)d_in[0];
    const int*   labels = (const int*)  d_in[1];
    const int*   tri    = (const int*)  d_in[2];
    const float* W1     = (const float*)d_in[3];
    const float* b1     = (const float*)d_in[4];
    const float* W2     = (const float*)d_in[5];
    const float* b2     = (const float*)d_in[6];
    const float* W3     = (const float*)d_in[7];
    const float* b3     = (const float*)d_in[8];
    float* out = (float*)d_out;

    const int M = BATCH * NPTS;
    char* ws = (char*)d_ws;
    size_t off = 0;
    auto carve = [&](size_t bytes) { char* p = ws + off; off += (bytes + 255) & ~(size_t)255; return p; };
    unsigned short* xbf  = (unsigned short*)carve((size_t)M * CIN * 2);
    unsigned short* w1bf = (unsigned short*)carve((size_t)256 * 128 * 2);
    unsigned short* w2bf = (unsigned short*)carve((size_t)256 * 256 * 2);
    unsigned short* w3bf = (unsigned short*)carve((size_t)64 * 256 * 2);
    unsigned short* h1   = (unsigned short*)carve((size_t)M * 256 * 2);
    unsigned short* h2   = (unsigned short*)carve((size_t)M * 256 * 2);
    unsigned short* ubf  = (unsigned short*)carve((size_t)M * 64 * 2);
    float*          sqn  = (float*)         carve((size_t)M * 4);
    unsigned char*  A0   = (unsigned char*) carve((size_t)BATCH * NPTS * NPTS);
    unsigned char*  A1   = (unsigned char*) carve((size_t)BATCH * NPTS * NPTS);
    int*            flags= (int*)           carve(256);
    int* changed = flags, *converged = flags + 1;

    cvt_bf16_kernel<<<(M * CIN + 255) / 256, 256, 0, stream>>>(x,  xbf,  M * CIN);
    cvt_bf16_kernel<<<(256 * 128 + 255) / 256, 256, 0, stream>>>(W1, w1bf, 256 * 128);
    cvt_bf16_kernel<<<(256 * 256 + 255) / 256, 256, 0, stream>>>(W2, w2bf, 256 * 256);
    cvt_bf16_kernel<<<(64 * 256 + 255) / 256, 256, 0, stream>>>(W3, w3bf, 64 * 256);
    init_kernel<<<1, 32, 0, stream>>>(flags, out + BATCH * NPTS);

    auto blocksFor = [](int tiles) { return (tiles * 32 + 255) / 256; };
    gemm_bf16_kernel<<<blocksFor((M / 16) * (256 / 16)), 256, 0, stream>>>(xbf, w1bf, b1, h1, M, 256, 128, 1);
    gemm_bf16_kernel<<<blocksFor((M / 16) * (256 / 16)), 256, 0, stream>>>(h1,  w2bf, b2, h2, M, 256, 256, 1);
    gemm_bf16_kernel<<<blocksFor((M / 16) * (64  / 16)), 256, 0, stream>>>(h2,  w3bf, b3, ubf, M, 64, 256, 0);

    rownorm_kernel<<<(M + 255) / 256, 256, 0, stream>>>(ubf, sqn, M);

    {
        int tiles = BATCH * (NPTS / 16) * (NPTS / 16);
        gram_adj_kernel<<<blocksFor(tiles), 256, 0, stream>>>(ubf, sqn, A0);
    }

    triplet_kernel<<<BATCH, 256, 0, stream>>>(labels, tri, ubf, sqn, out + BATCH * NPTS);

    unsigned char* bufs[2] = {A0, A1};
    const int nblocks = BATCH * (NPTS / 128) * (NPTS / 64);
    for (int it = 0; it < 11; ++it) {
        bool_square_kernel<<<nblocks, 256, 0, stream>>>(bufs[it & 1], bufs[(it & 1) ^ 1], changed, converged);
        closure_flag_kernel<<<1, 32, 0, stream>>>(changed, converged);
    }

    clusters_kernel<<<(BATCH * NPTS + 255) / 256, 256, 0, stream>>>(bufs[1], out);
}
